// TransformerBlock_55070070670104
// MI455X (gfx1250) — matrix operand order repaired and checked by reading, not compiled, not run on hardware
//
#include <hip/hip_runtime.h>
#include <hip/hip_bf16.h>
#include <math.h>

// ---------------------------------------------------------------------------
// Transformer block for MI455X (gfx1250, wave32, WMMA).
// GEMMs in bf16 via v_wmma_f32_16x16x32_bf16, f32 accumulate.
// Tile staging via GLOBAL_LOAD_ASYNC_TO_LDS_B128 (ASYNCcnt) + double buffering.
// ---------------------------------------------------------------------------

#define D_MODEL 2048
#define N_HEADS 16
#define HEAD_DIM 128
#define D_FF 5504
#define BB 2
#define TT 2048
#define MROWS (BB * TT)   // 4096
#define EPSV 1e-8f

typedef __attribute__((ext_vector_type(16))) __bf16 v16bf;
typedef __attribute__((ext_vector_type(8)))  float  v8f;
typedef __attribute__((ext_vector_type(4)))  int    i32x4;

union Frag16 { v16bf v; __bf16 e[16]; };

__device__ inline v8f wmma_bf16(v16bf a, v16bf b, v8f c) {
    // (neg_a, A, neg_b, B, c_mod, C, reuse_a, reuse_b)
    return __builtin_amdgcn_wmma_f32_16x16x32_bf16(false, a, false, b,
                                                   (short)0, c, false, false);
}

// ---- async global -> LDS copy (16B per lane), ASYNCcnt-tracked --------------
#if __has_builtin(__builtin_amdgcn_global_load_async_to_lds_b128)
#define ASYNC_LDS_COPY 1
#else
#define ASYNC_LDS_COPY 0
#endif

__device__ inline void cp16(const __bf16* g, __bf16* l) {
#if ASYNC_LDS_COPY
    __builtin_amdgcn_global_load_async_to_lds_b128(
        (__attribute__((address_space(1))) i32x4*)(uintptr_t)g,
        (__attribute__((address_space(3))) i32x4*)(uintptr_t)l,
        0, 0);
#else
    *(uint4*)l = *(const uint4*)g;
#endif
}

__device__ inline void cp_wait() {
#if ASYNC_LDS_COPY
#if __has_builtin(__builtin_amdgcn_s_wait_asynccnt)
    __builtin_amdgcn_s_wait_asynccnt(0);
#else
    asm volatile("s_wait_asynccnt 0x0" ::: "memory");
#endif
#endif
}

// A-matrix 16x32 bf16 fragment from row-major LDS tile (leading dim = ldk).
// lanes 0-15: K = {kk..kk+7, kk+16..kk+23}; lanes 16-31: +8 (ISA 7.12.2).
__device__ inline v16bf load_frag_a(const __bf16* lds, int row, int kk,
                                    int lane, int ldk) {
    const int off = (lane >= 16) ? 8 : 0;
    Frag16 f;
    const __bf16* p0 = lds + row * ldk + kk + off;
    const __bf16* p1 = lds + row * ldk + kk + 16 + off;
#pragma unroll
    for (int i = 0; i < 8; ++i) { f.e[i] = p0[i]; f.e[8 + i] = p1[i]; }
    return f.v;
}

// B-matrix 32x16 bf16 fragment. LDS tile stored [n][k] row-major (ldk).
// lanes 0-15: K = {kk..kk+7, kk+16..kk+23}; lanes 16-31: +8 (matches A).
__device__ inline v16bf load_frag_b(const __bf16* lds, int ncol, int kk,
                                    int lane, int ldk) {
    const int off = (lane >= 16) ? 8 : 0;
    Frag16 f;
    const __bf16* p = lds + ncol * ldk + kk + off;
#pragma unroll
    for (int i = 0; i < 8; ++i) { f.e[i] = p[i]; f.e[8 + i] = p[16 + i]; }
    return f.v;
}

// ---------------------------------------------------------------------------
// Elementwise / norm kernels
// ---------------------------------------------------------------------------

__global__ void cvt_f32_bf16_kernel(const float* __restrict__ in,
                                    __bf16* __restrict__ out, int n) {
    int i = blockIdx.x * blockDim.x + threadIdx.x;
    if (i < n) out[i] = (__bf16)in[i];
}

__global__ __launch_bounds__(256)
void rmsnorm_kernel(const float* __restrict__ x, const float* __restrict__ g,
                    __bf16* __restrict__ out) {
    __shared__ float red[256];
    const int row = blockIdx.x;
    const float* xr = x + (size_t)row * D_MODEL;
    float ss = 0.f;
    for (int i = threadIdx.x; i < D_MODEL; i += 256) {
        float v = xr[i];
        ss += v * v;
    }
    red[threadIdx.x] = ss;
    __syncthreads();
    for (int off = 128; off > 0; off >>= 1) {
        if (threadIdx.x < off) red[threadIdx.x] += red[threadIdx.x + off];
        __syncthreads();
    }
    const float rms = sqrtf(red[0] / (float)D_MODEL);
    const float inv = 1.f / (rms + EPSV);
    for (int i = threadIdx.x; i < D_MODEL; i += 256)
        out[(size_t)row * D_MODEL + i] = (__bf16)(g[i] * xr[i] * inv);
}

// In-place RoPE on a bf16 [B,T,H,HD] tensor; each thread owns a (d, d+64) pair.
__global__ void rope_kernel(__bf16* __restrict__ q,
                            const float* __restrict__ cosb,
                            const float* __restrict__ sinb, int n) {
    int idx = blockIdx.x * blockDim.x + threadIdx.x;
    if (idx >= n) return;
    const int d = idx & 63;
    const int h = (idx >> 6) & (N_HEADS - 1);
    const int t = (idx >> 10) & (TT - 1);
    const int b = idx >> 21;
    const float c = cosb[t * HEAD_DIM + d];
    const float s = sinb[t * HEAD_DIM + d];   // sin/cos repeat at d+64
    const size_t base = ((size_t)(b * TT + t) * D_MODEL) + h * HEAD_DIM;
    const float x0 = (float)q[base + d];
    const float x1 = (float)q[base + d + 64];
    q[base + d]      = (__bf16)(x0 * c - x1 * s);
    q[base + d + 64] = (__bf16)(x1 * c + x0 * s);
}

__global__ void silu_mul_kernel(const __bf16* __restrict__ a,
                                const __bf16* __restrict__ b,
                                __bf16* __restrict__ o, int n) {
    int i = blockIdx.x * blockDim.x + threadIdx.x;
    if (i < n) {
        float x = (float)a[i];
        float s = x / (1.f + __expf(-x));
        o[i] = (__bf16)(s * (float)b[i]);
    }
}

// ---------------------------------------------------------------------------
// GEMM: C[M,N] = A[M,K] (bf16) * W[N,K]^T (bf16), f32 accumulate.
// Block tile 128x128, K-step 32, 8 waves; wave tile 32x64 (8 WMMA tiles).
// Double-buffered LDS with async global->LDS staging.
// EPI: 0 = store f32, 1 = store bf16, 2 = store f32 (Res + acc)
// ---------------------------------------------------------------------------
template <int EPI>
__global__ __launch_bounds__(256)
void gemm_bf16_kernel(const __bf16* __restrict__ A, const __bf16* __restrict__ W,
                      void* __restrict__ Cout, const float* __restrict__ Res,
                      int M, int N, int K) {
    __shared__ __bf16 As[2][128 * 32];
    __shared__ __bf16 Ws[2][128 * 32];

    const int tid  = threadIdx.x;
    const int lane = tid & 31;
    const int wave = tid >> 5;
    const int m0 = blockIdx.y * 128;
    const int n0 = blockIdx.x * 128;
    const int rt = (wave & 3) * 32;        // wave row base within block tile
    const int ct = (wave >> 2) * 64;       // wave col base within block tile

    v8f acc[2][4] = {};

    const int lrow = tid >> 1;             // 0..127
    const int lcol = (tid & 1) * 16;       // 0 or 16

    const __bf16* agBase = A + (size_t)(m0 + lrow) * K + lcol;
    const __bf16* wgBase = W + (size_t)(n0 + lrow) * K + lcol;
    __bf16* asDst0 = &As[0][lrow * 32 + lcol];
    __bf16* asDst1 = &As[1][lrow * 32 + lcol];
    __bf16* wsDst0 = &Ws[0][lrow * 32 + lcol];
    __bf16* wsDst1 = &Ws[1][lrow * 32 + lcol];

    auto stage = [&](int bsel, int k0) {
        __bf16* ad = bsel ? asDst1 : asDst0;
        __bf16* wd = bsel ? wsDst1 : wsDst0;
        cp16(agBase + k0,     ad);
        cp16(agBase + k0 + 8, ad + 8);
        cp16(wgBase + k0,     wd);
        cp16(wgBase + k0 + 8, wd + 8);
    };

    stage(0, 0);
    cp_wait();
    __syncthreads();

    int buf = 0;
    for (int k0 = 0; k0 < K; k0 += 32) {
        if (k0 + 32 < K) stage(buf ^ 1, k0 + 32);   // prefetch next K tile

        const __bf16* as = As[buf];
        const __bf16* ws = Ws[buf];
        v16bf a0 = load_frag_a(as, rt + (lane & 15), 0, lane, 32);
        v16bf a1 = load_frag_a(as, rt + 16 + (lane & 15), 0, lane, 32);
#pragma unroll
        for (int j = 0; j < 4; ++j) {
            v16bf bfr = load_frag_b(ws, ct + j * 16 + (lane & 15), 0, lane, 32);
            acc[0][j] = wmma_bf16(a0, bfr, acc[0][j]);
            acc[1][j] = wmma_bf16(a1, bfr, acc[1][j]);
        }
        cp_wait();
        __syncthreads();
        buf ^= 1;
    }

    // Epilogue: C layout -> col = lane&15, row = vgpr + (lane>=16 ? 8 : 0).
#pragma unroll
    for (int i = 0; i < 2; ++i) {
        const int rr = m0 + rt + i * 16 + (lane >> 4) * 8;
#pragma unroll
        for (int j = 0; j < 4; ++j) {
            const int c = n0 + ct + j * 16 + (lane & 15);
#pragma unroll
            for (int v = 0; v < 8; ++v) {
                const size_t idx = (size_t)(rr + v) * N + c;
                const float val = acc[i][j][v];
                if (EPI == 0)       ((float*)Cout)[idx]  = val;
                else if (EPI == 1)  ((__bf16*)Cout)[idx] = (__bf16)val;
                else                ((float*)Cout)[idx]  = Res[idx] + val;
            }
        }
    }
}

// ---------------------------------------------------------------------------
// Flash-style causal attention. One workgroup per (b, h, 64-query block).
// Q tile resident in LDS; stream 64-key K/V tiles. WMMA for QK^T and P*V.
// ---------------------------------------------------------------------------
__global__ __launch_bounds__(256)
void attention_kernel(const __bf16* __restrict__ Q, const __bf16* __restrict__ K,
                      const __bf16* __restrict__ V, __bf16* __restrict__ O) {
    __shared__ __bf16 Qs[64 * HEAD_DIM];   // 16 KB
    __shared__ __bf16 KVs[64 * HEAD_DIM];  // 16 KB (K tile, then V^T tile)
    __shared__ float  Ss[64 * 64];         // 16 KB
    __shared__ __bf16 Ps[64 * 64];         //  8 KB
    __shared__ float  mrow[64], lrow[64], crow[64];

    const int tid  = threadIdx.x;
    const int lane = tid & 31;
    const int wave = tid >> 5;
    const int nqb = TT / 64;
    const int qb = blockIdx.x % nqb;
    const int h  = (blockIdx.x / nqb) % N_HEADS;
    const int b  = blockIdx.x / (nqb * N_HEADS);
    const int q0 = qb * 64;

    // Load Q tile via async copies (64B per thread)
    {
        const int r = tid >> 2;
        const int c = (tid & 3) * 32;
        const __bf16* src = Q + ((size_t)(b * TT + q0 + r) * D_MODEL + h * HEAD_DIM + c);
        __bf16* dst = Qs + r * HEAD_DIM + c;
        cp16(src, dst);
        cp16(src + 8,  dst + 8);
        cp16(src + 16, dst + 16);
        cp16(src + 24, dst + 24);
    }
    if (tid < 64) { mrow[tid] = -3.0e38f; lrow[tid] = 0.f; }

    const int rt = wave & 3;               // O/S row-tile (q rows 16*rt..)
    const int cw = (wave >> 2) * 4;        // O col-tile base (of 8)
    v8f oacc[4] = {};
    const float scale = 0.08838834764831845f;  // 1/sqrt(128)

    cp_wait();
    __syncthreads();

    for (int jb = 0; jb <= qb; ++jb) {
        const int k0 = jb * 64;

        // Load K tile (async global -> LDS)
        {
            const int r = tid >> 2;
            const int c = (tid & 3) * 32;
            const __bf16* src = K + ((size_t)(b * TT + k0 + r) * D_MODEL + h * HEAD_DIM + c);
            __bf16* dst = KVs + r * HEAD_DIM + c;
            cp16(src, dst);
            cp16(src + 8,  dst + 8);
            cp16(src + 16, dst + 16);
            cp16(src + 24, dst + 24);
        }
        cp_wait();
        __syncthreads();

        // S = scale * Q K^T  (wave: row-tile rt, 2 col-tiles)
        {
            const int ctb = (wave >> 2) * 2;
            v8f s0 = {}, s1 = {};
#pragma unroll
            for (int kk = 0; kk < HEAD_DIM; kk += 32) {
                v16bf a  = load_frag_a(Qs, rt * 16 + (lane & 15), kk, lane, HEAD_DIM);
                v16bf b0 = load_frag_b(KVs, ctb * 16 + (lane & 15), kk, lane, HEAD_DIM);
                v16bf b1 = load_frag_b(KVs, (ctb + 1) * 16 + (lane & 15), kk, lane, HEAD_DIM);
                s0 = wmma_bf16(a, b0, s0);
                s1 = wmma_bf16(a, b1, s1);
            }
            const int rr = rt * 16 + (lane >> 4) * 8;
            const int c0 = ctb * 16 + (lane & 15);
#pragma unroll
            for (int v = 0; v < 8; ++v) {
                Ss[(rr + v) * 64 + c0]      = s0[v] * scale;
                Ss[(rr + v) * 64 + c0 + 16] = s1[v] * scale;
            }
        }
        __syncthreads();

        // Online softmax, one row per thread (threads 0..63); causal mask.
        if (tid < 64) {
            const int r = tid;
            const int kmax = q0 + r - k0;      // valid columns c <= kmax
            float m = mrow[r];
            for (int c = 0; c < 64; ++c)
                if (c <= kmax) m = fmaxf(m, Ss[r * 64 + c]);
            const float cr = __expf(mrow[r] - m);
            float sum = 0.f;
            for (int c = 0; c < 64; ++c) {
                float p = (c <= kmax) ? __expf(Ss[r * 64 + c] - m) : 0.f;
                Ps[r * 64 + c] = (__bf16)p;
                sum += p;
            }
            lrow[r] = lrow[r] * cr + sum;
            mrow[r] = m;
            crow[r] = cr;
        }
        // Load V tile transposed into KVs (Vt[d][t_k]); K tile fully consumed.
        {
            const int r  = tid >> 2;
            const int c0 = (tid & 3) * 32;
            const __bf16* src = V + ((size_t)(b * TT + k0 + r) * D_MODEL + h * HEAD_DIM + c0);
            for (int i = 0; i < 32; ++i)
                KVs[(c0 + i) * 64 + r] = src[i];
        }
        __syncthreads();

        // O = O * corr + P * V
        {
            const int rr = rt * 16 + (lane >> 4) * 8;
            float cf[8];
#pragma unroll
            for (int v = 0; v < 8; ++v) cf[v] = crow[rr + v];
#pragma unroll
            for (int i = 0; i < 4; ++i)
#pragma unroll
                for (int v = 0; v < 8; ++v) oacc[i][v] *= cf[v];
        }
#pragma unroll
        for (int kk = 0; kk < 64; kk += 32) {
            v16bf a = load_frag_a(Ps, rt * 16 + (lane & 15), kk, lane, 64);
#pragma unroll
            for (int i = 0; i < 4; ++i) {
                v16bf bf = load_frag_b(KVs, (cw + i) * 16 + (lane & 15), kk, lane, 64);
                oacc[i] = wmma_bf16(a, bf, oacc[i]);
            }
        }
        __syncthreads();
    }

    // Epilogue: O / l -> bf16 [b,t,h,d]
    {
        const int rr = rt * 16 + (lane >> 4) * 8;
        float linv[8];
#pragma unroll
        for (int v = 0; v < 8; ++v) linv[v] = 1.f / lrow[rr + v];
#pragma unroll
        for (int i = 0; i < 4; ++i) {
            const int c = (cw + i) * 16 + (lane & 15);
#pragma unroll
            for (int v = 0; v < 8; ++v) {
                const size_t idx =
                    (size_t)(b * TT + q0 + rr + v) * D_MODEL + h * HEAD_DIM + c;
                O[idx] = (__bf16)(oacc[i][v] * linv[v]);
            }
        }
    }
}

// ---------------------------------------------------------------------------
// Host orchestration
// ---------------------------------------------------------------------------
extern "C" void kernel_launch(void* const* d_in, const int* in_sizes, int n_in,
                              void* d_out, int out_size, void* d_ws, size_t ws_size,
                              hipStream_t stream) {
    (void)in_sizes; (void)n_in; (void)out_size; (void)ws_size;

    const float* x    = (const float*)d_in[0];
    const float* cosb = (const float*)d_in[1];
    const float* sinb = (const float*)d_in[2];
    /* d_in[3] = mask : causal mask applied analytically in-kernel */
    const float* g1   = (const float*)d_in[4];
    const float* g2   = (const float*)d_in[5];
    const float* Wq   = (const float*)d_in[6];
    const float* Wk   = (const float*)d_in[7];
    const float* Wv   = (const float*)d_in[8];
    const float* Wo   = (const float*)d_in[9];
    const float* W1   = (const float*)d_in[10];
    const float* W2   = (const float*)d_in[11];
    const float* Wout = (const float*)d_in[12];
    float* out = (float*)d_out;

    char* p = (char*)d_ws;
    auto alloc = [&](size_t bytes) -> char* {
        char* r = p;
        p += (bytes + 255) & ~(size_t)255;
        return r;
    };

    const size_t DD   = (size_t)D_MODEL * D_MODEL;   // 4.19M
    const size_t DFD  = (size_t)D_FF * D_MODEL;      // 11.27M
    const size_t MD   = (size_t)MROWS * D_MODEL;     // 8.39M
    const size_t MF   = (size_t)MROWS * D_FF;        // 22.5M

    __bf16* wq_b   = (__bf16*)alloc(DD * 2);
    __bf16* wk_b   = (__bf16*)alloc(DD * 2);
    __bf16* wv_b   = (__bf16*)alloc(DD * 2);
    __bf16* wo_b   = (__bf16*)alloc(DD * 2);
    __bf16* w1_b   = (__bf16*)alloc(DFD * 2);
    __bf16* w2_b   = (__bf16*)alloc(DFD * 2);
    __bf16* wout_b = (__bf16*)alloc(DFD * 2);
    __bf16* h_b    = (__bf16*)alloc(MD * 2);
    __bf16* q_b    = (__bf16*)alloc(MD * 2);
    __bf16* k_b    = (__bf16*)alloc(MD * 2);
    __bf16* v_b    = (__bf16*)alloc(MD * 2);
    __bf16* attn_b = (__bf16*)alloc(MD * 2);
    float*  x1_f   = (float*)alloc(MD * 4);
    __bf16* h2_b   = (__bf16*)alloc(MD * 2);
    __bf16* ff1_b  = (__bf16*)alloc(MF * 2);
    __bf16* ff2_b  = (__bf16*)alloc(MF * 2);
    __bf16* g_b    = (__bf16*)alloc(MF * 2);

    auto cvt = [&](const float* src, __bf16* dst, size_t n) {
        cvt_f32_bf16_kernel<<<(unsigned)((n + 255) / 256), 256, 0, stream>>>(src, dst, (int)n);
    };
    cvt(Wq, wq_b, DD);   cvt(Wk, wk_b, DD);   cvt(Wv, wv_b, DD);
    cvt(Wo, wo_b, DD);   cvt(W1, w1_b, DFD);  cvt(W2, w2_b, DFD);
    cvt(Wout, wout_b, DFD);

    // h = rmsnorm(x, g1)
    rmsnorm_kernel<<<MROWS, 256, 0, stream>>>(x, g1, h_b);

    // Q,K,V projections (bf16 out)
    dim3 gP(D_MODEL / 128, MROWS / 128);
    gemm_bf16_kernel<1><<<gP, 256, 0, stream>>>(h_b, wq_b, q_b, nullptr, MROWS, D_MODEL, D_MODEL);
    gemm_bf16_kernel<1><<<gP, 256, 0, stream>>>(h_b, wk_b, k_b, nullptr, MROWS, D_MODEL, D_MODEL);
    gemm_bf16_kernel<1><<<gP, 256, 0, stream>>>(h_b, wv_b, v_b, nullptr, MROWS, D_MODEL, D_MODEL);

    // RoPE (in place)
    const int nrope = BB * TT * N_HEADS * 64;
    rope_kernel<<<nrope / 256, 256, 0, stream>>>(q_b, cosb, sinb, nrope);
    rope_kernel<<<nrope / 256, 256, 0, stream>>>(k_b, cosb, sinb, nrope);

    // Flash attention
    attention_kernel<<<BB * N_HEADS * (TT / 64), 256, 0, stream>>>(q_b, k_b, v_b, attn_b);

    // x1 = x + attn @ Wo^T
    gemm_bf16_kernel<2><<<gP, 256, 0, stream>>>(attn_b, wo_b, x1_f, x, MROWS, D_MODEL, D_MODEL);

    // h2 = rmsnorm(x1, g2)
    rmsnorm_kernel<<<MROWS, 256, 0, stream>>>(x1_f, g2, h2_b);

    // FFN
    dim3 gF(D_FF / 128, MROWS / 128);
    gemm_bf16_kernel<1><<<gF, 256, 0, stream>>>(h2_b, w1_b, ff1_b, nullptr, MROWS, D_FF, D_MODEL);
    gemm_bf16_kernel<1><<<gF, 256, 0, stream>>>(h2_b, w2_b, ff2_b, nullptr, MROWS, D_FF, D_MODEL);
    silu_mul_kernel<<<(unsigned)((MF + 255) / 256), 256, 0, stream>>>(ff1_b, ff2_b, g_b, (int)MF);

    // out = x1 + g @ Wout^T
    gemm_bf16_kernel<2><<<gP, 256, 0, stream>>>(g_b, wout_b, out, x1_f, MROWS, D_MODEL, D_FF);
}
